// Ernie4DecoderLayer_19353122635830
// MI455X (gfx1250) — compile-verified, matrix operand order checked by reading, not run on hardware
//
#include <hip/hip_runtime.h>
#include <hip/hip_bf16.h>
#include <math.h>

// ---------------------------------------------------------------------------
// Ernie4 decoder layer for MI455X (gfx1250, wave32, WMMA 16x16x32 f16)
// ---------------------------------------------------------------------------

typedef __attribute__((ext_vector_type(16))) _Float16 v16h;
typedef __attribute__((ext_vector_type(8)))  _Float16 v8h;
typedef __attribute__((ext_vector_type(8)))  float    v8f;
typedef __attribute__((ext_vector_type(4)))  float    v4f;

#define T_TOK   1024
#define HDIM    2048
#define NHEAD   16
#define NKVH    4
#define DHEAD   128
#define NEXP    16
#define IEXP    1024
#define ISHARED 2048

__device__ __forceinline__ v8f vzero8() {
  v8f z = {0.f, 0.f, 0.f, 0.f, 0.f, 0.f, 0.f, 0.f};
  return z;
}

// Fragment loader for V_WMMA_F32_16X16X32_F16 with the matrix stored as
// [row][ld] halves; per-lane K pattern per ISA 7.12.2:
//   lane <16 : row = r0+lane,    K = {0..7, 16..23}
//   lane>=16 : row = r0+lane-16, K = {8..15, 24..31}
__device__ __forceinline__ v16h frag_ld_k16(const _Float16* __restrict__ base,
                                            int ld, int r0) {
  int lane = threadIdx.x & 31;
  const _Float16* p = base + (size_t)(r0 + (lane & 15)) * ld + ((lane >> 4) << 3);
  v8h lo = *(const v8h*)(p);
  v8h hi = *(const v8h*)(p + 16);
  v16h r;
#pragma unroll
  for (int i = 0; i < 8; ++i) { r[i] = lo[i]; r[i + 8] = hi[i]; }
  return r;
}

// B-fragment where the matrix is stored K-major: element (k, n) at base[k*ldk + n].
__device__ __forceinline__ v16h frag_ld_b_kmajor(const _Float16* __restrict__ base,
                                                 int ldk, int n0) {
  int lane = threadIdx.x & 31;
  const _Float16* p = base + (n0 + (lane & 15)) + (size_t)((lane >> 4) << 3) * ldk;
  v16h r;
#pragma unroll
  for (int i = 0; i < 8; ++i) {
    r[i]     = p[(size_t)i * ldk];
    r[i + 8] = p[(size_t)(i + 16) * ldk];
  }
  return r;
}

// ---------------------------------------------------------------------------
// Workhorse GEMM: C[M][N] = epilogue( A_f16[M][K] * B_f32[K][N] )
// 128x128 block tile, BK=32, 8 waves, each wave a 32x64 sub-tile (2x4 WMMAs).
// Double-buffered LDS pipeline: next tile's global loads are issued before the
// single barrier, the current buffer feeds 8 WMMAs while loads are in flight,
// then registers are deposited into the alternate buffer. fp32 weights are
// converted to f16 while staging (transposed to [n][k] so per-lane K runs are
// contiguous); the 192MB L2 absorbs tile re-reads. global_prefetch issued two
// tiles ahead. Epilogue is compile-time specialized (branch-free).
// ---------------------------------------------------------------------------
#define BM 128
#define BN 128
#define BKT 32

template <bool ADD, bool SCALE, bool ACCUM>
__global__ __launch_bounds__(256) void gemm_xw_kernel(
    const _Float16* __restrict__ A, const float* __restrict__ B,
    float* __restrict__ C, int M, int N, int K,
    const float* __restrict__ addsrc, const float* __restrict__ rowscale,
    int rs_stride)
{
  __shared__ __align__(16) _Float16 As[2][BM][BKT];   // [buf][m][k]
  __shared__ __align__(16) _Float16 Bs[2][BN][BKT];   // [buf][n][k] (transposed)

  const int tid = threadIdx.x;
  const int wave = tid >> 5;
  const int lane = tid & 31;
  const int bm0 = blockIdx.y * BM;
  const int bn0 = blockIdx.x * BN;
  const int wr = wave & 3;    // 32-row group
  const int wc = wave >> 2;   // 64-col group

  v8f acc[2][4];
#pragma unroll
  for (int i = 0; i < 2; ++i)
#pragma unroll
    for (int j = 0; j < 4; ++j) acc[i][j] = vzero8();

  const int ar = tid >> 1;          // A staging: row 0..127
  const int ac = (tid & 1) << 4;    //            col 0 or 16
  const int bk = tid >> 3;          // B staging: k 0..31
  const int bn = (tid & 7) << 4;    //            n base 0..112

  const _Float16* gA = A + (size_t)(bm0 + ar) * K + ac;
  const float*    gB = B + (size_t)bk * N + bn0 + bn;

  // stage tile 0
  v8h ra0 = *(const v8h*)(gA);
  v8h ra1 = *(const v8h*)(gA + 8);
  v4f rb[4];
#pragma unroll
  for (int i = 0; i < 4; ++i) rb[i] = *(const v4f*)(gB + 4 * i);

  *(v8h*)&As[0][ar][ac]     = ra0;
  *(v8h*)&As[0][ar][ac + 8] = ra1;
#pragma unroll
  for (int i = 0; i < 4; ++i) {
    Bs[0][bn + 4 * i + 0][bk] = (_Float16)rb[i].x;
    Bs[0][bn + 4 * i + 1][bk] = (_Float16)rb[i].y;
    Bs[0][bn + 4 * i + 2][bk] = (_Float16)rb[i].z;
    Bs[0][bn + 4 * i + 3][bk] = (_Float16)rb[i].w;
  }

  for (int kt = 0; kt < K; kt += BKT) {
    const int cur = (kt / BKT) & 1;
    const int nxt = cur ^ 1;
    const bool more = (kt + BKT) < K;

    // issue next tile's global loads (latency overlapped with WMMAs below)
    if (more) {
      ra0 = *(const v8h*)(gA + kt + BKT);
      ra1 = *(const v8h*)(gA + kt + BKT + 8);
#pragma unroll
      for (int i = 0; i < 4; ++i)
        rb[i] = *(const v4f*)(gB + (size_t)(kt + BKT) * N + 4 * i);
    }
    // prefetch two tiles ahead into L2 (gfx1250 global_prefetch path)
    if (kt + 2 * BKT < K) {
      __builtin_prefetch(gA + kt + 2 * BKT, 0, 1);
      __builtin_prefetch(gB + (size_t)(kt + 2 * BKT) * N, 0, 1);
    }

    __syncthreads();   // buf[cur] stores (from previous iteration) visible

    v16h bfr[4];
#pragma unroll
    for (int j = 0; j < 4; ++j)
      bfr[j] = frag_ld_k16(&Bs[cur][0][0], BKT, wc * 64 + j * 16);
#pragma unroll
    for (int i = 0; i < 2; ++i) {
      v16h af = frag_ld_k16(&As[cur][0][0], BKT, wr * 32 + i * 16);
#pragma unroll
      for (int j = 0; j < 4; ++j)
        acc[i][j] = __builtin_amdgcn_wmma_f32_16x16x32_f16(
            false, af, false, bfr[j], (short)0, acc[i][j], false, false);
    }

    // deposit next tile into the alternate buffer (no hazard with buf[cur])
    if (more) {
      *(v8h*)&As[nxt][ar][ac]     = ra0;
      *(v8h*)&As[nxt][ar][ac + 8] = ra1;
#pragma unroll
      for (int i = 0; i < 4; ++i) {
        Bs[nxt][bn + 4 * i + 0][bk] = (_Float16)rb[i].x;
        Bs[nxt][bn + 4 * i + 1][bk] = (_Float16)rb[i].y;
        Bs[nxt][bn + 4 * i + 2][bk] = (_Float16)rb[i].z;
        Bs[nxt][bn + 4 * i + 3][bk] = (_Float16)rb[i].w;
      }
    }
  }

  // epilogue (C/D layout: VGPR r -> M=r (lanes 0-15) / M=r+8 (lanes 16-31))
  const int cno = lane & 15;
  const int rbase = (lane >> 4) << 3;
#pragma unroll
  for (int i = 0; i < 2; ++i)
#pragma unroll
    for (int j = 0; j < 4; ++j) {
      const int mt = bm0 + wr * 32 + i * 16 + rbase;
      const int n  = bn0 + wc * 64 + j * 16 + cno;
#pragma unroll
      for (int r = 0; r < 8; ++r) {
        const int m = mt + r;
        float v = acc[i][j][r];
        if constexpr (SCALE) v *= rowscale[(size_t)m * rs_stride];
        float base = 0.f;
        if constexpr (ACCUM) base = C[(size_t)m * N + n];
        if constexpr (ADD)   base += addsrc[(size_t)m * N + n];
        C[(size_t)m * N + n] = base + v;
      }
    }
}

// ---------------------------------------------------------------------------
// RMSNorm (one token per block), writes f16 (WMMA operand) and optional f32.
// ---------------------------------------------------------------------------
__global__ __launch_bounds__(256) void rmsnorm_kernel(
    const float* __restrict__ x, const float* __restrict__ w,
    _Float16* __restrict__ out16, float* __restrict__ out32)
{
  __shared__ float red[8];
  const int t = blockIdx.x;
  const float* row = x + (size_t)t * HDIM;
  float ss = 0.f;
  for (int c = threadIdx.x; c < HDIM; c += 256) { float v = row[c]; ss += v * v; }
#pragma unroll
  for (int off = 16; off > 0; off >>= 1) ss += __shfl_down(ss, off, 32);
  if ((threadIdx.x & 31) == 0) red[threadIdx.x >> 5] = ss;
  __syncthreads();
  if (threadIdx.x == 0) {
    float s = 0.f;
    for (int i = 0; i < 8; ++i) s += red[i];
    red[0] = rsqrtf(s / (float)HDIM + 1e-6f);
  }
  __syncthreads();
  const float r = red[0];
  for (int c = threadIdx.x; c < HDIM; c += 256) {
    float v = row[c] * r * w[c];
    out16[(size_t)t * HDIM + c] = (_Float16)v;
    if (out32) out32[(size_t)t * HDIM + c] = v;
  }
}

// ---------------------------------------------------------------------------
// RoPE on q/k (interleaved pairs, matches reference) + f16 convert of q/k/v.
// ---------------------------------------------------------------------------
__global__ __launch_bounds__(256) void rope_cvt_kernel(
    const int* __restrict__ pos,
    const float* __restrict__ q, const float* __restrict__ k,
    const float* __restrict__ v,
    _Float16* __restrict__ q16, _Float16* __restrict__ k16,
    _Float16* __restrict__ v16)
{
  const int t = blockIdx.x;
  const float p = (float)pos[t];

  for (int i = threadIdx.x; i < NHEAD * DHEAD / 2; i += 256) {
    const int head = i / (DHEAD / 2);
    const int j = i % (DHEAD / 2);
    const float inv = __powf(10000.0f, -(float)(2 * j) / (float)DHEAD);
    float sn, cs;
    __sincosf(p * inv, &sn, &cs);
    const size_t base = (size_t)t * NHEAD * DHEAD + (size_t)head * DHEAD + 2 * j;
    const float x1 = q[base], x2 = q[base + 1];
    q16[base]     = (_Float16)(x1 * cs - x2 * sn);
    q16[base + 1] = (_Float16)(x1 * sn + x2 * cs);
  }
  for (int i = threadIdx.x; i < NKVH * DHEAD / 2; i += 256) {
    const int head = i / (DHEAD / 2);
    const int j = i % (DHEAD / 2);
    const float inv = __powf(10000.0f, -(float)(2 * j) / (float)DHEAD);
    float sn, cs;
    __sincosf(p * inv, &sn, &cs);
    const size_t base = (size_t)t * NKVH * DHEAD + (size_t)head * DHEAD + 2 * j;
    const float x1 = k[base], x2 = k[base + 1];
    k16[base]     = (_Float16)(x1 * cs - x2 * sn);
    k16[base + 1] = (_Float16)(x1 * sn + x2 * cs);
  }
  for (int i = threadIdx.x; i < NKVH * DHEAD; i += 256) {
    const size_t base = (size_t)t * NKVH * DHEAD + i;
    v16[base] = (_Float16)v[base];
  }
}

// ---------------------------------------------------------------------------
// Fused causal attention, one (head, 16-query block) per workgroup, 8 waves.
// Dynamic LDS: scores f32[16][1024] + exp f16[16][1024] + reductions (~97 KB,
// legal within the 320 KB/WGP CDNA5 budget).
// ---------------------------------------------------------------------------
__global__ __launch_bounds__(256) void attn_kernel(
    const _Float16* __restrict__ qf, const _Float16* __restrict__ kf,
    const _Float16* __restrict__ vf, _Float16* __restrict__ of)
{
  extern __shared__ char smem[];
  float    (*sS)[T_TOK] = (float    (*)[T_TOK])(smem);
  _Float16 (*sP)[T_TOK] = (_Float16 (*)[T_TOK])(smem + (size_t)16 * T_TOK * 4);
  float* rowred = (float*)(smem + (size_t)16 * T_TOK * 6);
  float* rowsum = rowred + 256;

  const int tid = threadIdx.x;
  const int wave = tid >> 5;
  const int lane = tid & 31;
  const int qb = blockIdx.x;            // query block (16 rows)
  const int h  = blockIdx.y;            // head
  const int kvh = h >> 2;               // GQA: NH/NKV = 4
  const int q0 = qb * 16;

  // preload the 4 Q K-step fragments (D = 128 = 4 x 32)
  const _Float16* qbase = qf + (size_t)q0 * (NHEAD * DHEAD) + (size_t)h * DHEAD;
  v16h aq[4];
#pragma unroll
  for (int d = 0; d < 4; ++d) aq[d] = frag_ld_k16(qbase + d * 32, NHEAD * DHEAD, 0);

  const float scale = 0.08838834764831845f;  // 1/sqrt(128)

  // phase 1: S = Q K^T  (each wave handles key tiles wave, wave+8, ...)
  for (int kt = wave; kt < T_TOK / 16; kt += 8) {
    const int k0 = kt * 16;
    const _Float16* kb = kf + (size_t)k0 * (NKVH * DHEAD) + (size_t)kvh * DHEAD;
    v8f c = vzero8();
#pragma unroll
    for (int d = 0; d < 4; ++d) {
      v16h bk = frag_ld_k16(kb + d * 32, NKVH * DHEAD, 0);
      c = __builtin_amdgcn_wmma_f32_16x16x32_f16(false, aq[d], false, bk,
                                                 (short)0, c, false, false);
    }
    const int kn = k0 + (lane & 15);
    const int rb = (lane >> 4) << 3;
#pragma unroll
    for (int r = 0; r < 8; ++r) {
      const int qr = rb + r;
      float v = c[r] * scale;
      if (kn > q0 + qr) v = -1.0e9f;   // causal mask (matches reference)
      sS[qr][kn] = v;
    }
  }
  __syncthreads();

  // phase 2: softmax rows (store unnormalized exp in f16, keep row sums)
  {
    const int r = tid >> 4;
    const int sub = tid & 15;
    float mx = -3.0e38f;
    for (int c = sub; c < T_TOK; c += 16) mx = fmaxf(mx, sS[r][c]);
    rowred[r * 16 + sub] = mx;
    __syncthreads();
    if (sub == 0) {
      float m = rowred[r * 16];
      for (int i = 1; i < 16; ++i) m = fmaxf(m, rowred[r * 16 + i]);
      rowred[r * 16] = m;
    }
    __syncthreads();
    const float m = rowred[r * 16];
    float s = 0.f;
    for (int c = sub; c < T_TOK; c += 16) {
      float e = __expf(sS[r][c] - m);
      sP[r][c] = (_Float16)e;
      s += e;
    }
    __syncthreads();
    rowred[r * 16 + sub] = s;
    __syncthreads();
    if (sub == 0) {
      float t2 = 0.f;
      for (int i = 0; i < 16; ++i) t2 += rowred[r * 16 + i];
      rowsum[r] = t2;
    }
    __syncthreads();
  }

  // phase 3: O = P V  (wave w owns D-tile w; K-dim = 1024 keys = 32 steps)
  {
    const int n0 = wave * 16;
    v8f c = vzero8();
    const _Float16* vb = vf + (size_t)kvh * DHEAD;
    for (int k0 = 0; k0 < T_TOK; k0 += 32) {
      v16h ap = frag_ld_k16(&sP[0][0] + k0, T_TOK, 0);
      v16h bv = frag_ld_b_kmajor(vb + (size_t)k0 * (NKVH * DHEAD),
                                 NKVH * DHEAD, n0);
      c = __builtin_amdgcn_wmma_f32_16x16x32_f16(false, ap, false, bv,
                                                 (short)0, c, false, false);
    }
    const int dn = n0 + (lane & 15);
    const int rb = (lane >> 4) << 3;
#pragma unroll
    for (int r = 0; r < 8; ++r) {
      const int qr = rb + r;
      const float o = c[r] / rowsum[qr];
      of[(size_t)(q0 + qr) * (NHEAD * DHEAD) + (size_t)h * DHEAD + dn] = (_Float16)o;
    }
  }
}

// ---------------------------------------------------------------------------
// silu(g) * u -> f16 (next GEMM's A operand)
// ---------------------------------------------------------------------------
__global__ void silu_mul_kernel(const float* __restrict__ g,
                                const float* __restrict__ u,
                                _Float16* __restrict__ out, int n)
{
  const int i = blockIdx.x * blockDim.x + threadIdx.x;
  if (i < n) {
    const float gv = g[i];
    const float s = gv / (1.f + __expf(-gv));
    out[i] = (_Float16)(s * u[i]);
  }
}

// ---------------------------------------------------------------------------
// Router: logits -> sigmoid -> top-2 of (score+bias) -> normalized dense
// per-token expert weights wed[T][E] (zero where unselected).
// ---------------------------------------------------------------------------
__global__ __launch_bounds__(256) void router_kernel(
    const float* __restrict__ x, const float* __restrict__ gw,
    const float* __restrict__ gb, float* __restrict__ wed)
{
  __shared__ float part[16][16];  // [chunk][expert]
  __shared__ float sc[16];
  const int t = blockIdx.x;
  const int e = threadIdx.x & 15;
  const int ch = threadIdx.x >> 4;
  const float* row = x + (size_t)t * HDIM;
  const float* w = gw + (size_t)e * HDIM;
  float s = 0.f;
  for (int c = ch * 128; c < ch * 128 + 128; ++c) s += row[c] * w[c];
  part[ch][e] = s;
  __syncthreads();
  if (threadIdx.x < 16) {
    float l = 0.f;
    for (int i = 0; i < 16; ++i) l += part[i][threadIdx.x];
    sc[threadIdx.x] = 1.f / (1.f + __expf(-l));
  }
  __syncthreads();
  if (threadIdx.x == 0) {
    float a0 = -1e30f, a1 = -1e30f;
    int i0 = 0, i1 = 0;
    for (int i = 0; i < 16; ++i) {
      const float v = sc[i] + gb[i];
      if (v > a0)      { a1 = a0; i1 = i0; a0 = v; i0 = i; }
      else if (v > a1) { a1 = v; i1 = i; }
    }
    const float w0 = sc[i0], w1 = sc[i1];
    const float inv = 1.f / (w0 + w1);
    for (int i = 0; i < 16; ++i) wed[(size_t)t * 16 + i] = 0.f;
    wed[(size_t)t * 16 + i0] = w0 * inv;
    wed[(size_t)t * 16 + i1] = w1 * inv;
  }
}

// ---------------------------------------------------------------------------
// Host orchestration
// ---------------------------------------------------------------------------
static inline void launch_gemm(dim3 grid, hipStream_t st,
                               const _Float16* A, const float* B, float* C,
                               int M, int N, int K,
                               const float* addsrc, const float* rowscale,
                               int rs_stride)
{
  const dim3 blk(256);
  if (addsrc)
    gemm_xw_kernel<true, false, false><<<grid, blk, 0, st>>>(
        A, B, C, M, N, K, addsrc, nullptr, 0);
  else if (rowscale)
    gemm_xw_kernel<false, true, true><<<grid, blk, 0, st>>>(
        A, B, C, M, N, K, nullptr, rowscale, rs_stride);
  else
    gemm_xw_kernel<false, false, false><<<grid, blk, 0, st>>>(
        A, B, C, M, N, K, nullptr, nullptr, 0);
}

extern "C" void kernel_launch(void* const* d_in, const int* in_sizes, int n_in,
                              void* d_out, int out_size, void* d_ws, size_t ws_size,
                              hipStream_t stream)
{
  (void)in_sizes; (void)n_in; (void)out_size; (void)ws_size;

  const int*   positions = (const int*)d_in[0];
  const float* hidden    = (const float*)d_in[1];
  const float* w_in_ln   = (const float*)d_in[2];
  const float* w_post_ln = (const float*)d_in[3];
  const float* wq        = (const float*)d_in[4];
  const float* wk        = (const float*)d_in[5];
  const float* wv        = (const float*)d_in[6];
  const float* wo        = (const float*)d_in[7];
  const float* gate_w    = (const float*)d_in[8];
  const float* gate_b    = (const float*)d_in[9];
  const float* we_gate   = (const float*)d_in[10];
  const float* we_up     = (const float*)d_in[11];
  const float* we_down   = (const float*)d_in[12];
  const float* ws_gate   = (const float*)d_in[13];
  const float* ws_up     = (const float*)d_in[14];
  const float* ws_down   = (const float*)d_in[15];

  float* out   = (float*)d_out;                               // [T][H]
  float* resid = (float*)d_out + (size_t)T_TOK * HDIM;        // [T][H]

  char* ws = (char*)d_ws;
  size_t off = 0;
  auto carve = [&](size_t bytes) -> char* {
    char* p = ws + off;
    off = (off + bytes + 255) & ~(size_t)255;
    return p;
  };

  _Float16* x1h  = (_Float16*)carve((size_t)T_TOK * HDIM * 2);
  float*    qf32 = (float*)   carve((size_t)T_TOK * NHEAD * DHEAD * 4);
  float*    kf32 = (float*)   carve((size_t)T_TOK * NKVH * DHEAD * 4);
  float*    vf32 = (float*)   carve((size_t)T_TOK * NKVH * DHEAD * 4);
  _Float16* q16  = (_Float16*)carve((size_t)T_TOK * NHEAD * DHEAD * 2);
  _Float16* k16  = (_Float16*)carve((size_t)T_TOK * NKVH * DHEAD * 2);
  _Float16* v16  = (_Float16*)carve((size_t)T_TOK * NKVH * DHEAD * 2);
  _Float16* o16  = (_Float16*)carve((size_t)T_TOK * NHEAD * DHEAD * 2);
  _Float16* x2h  = (_Float16*)carve((size_t)T_TOK * HDIM * 2);
  float*    x2f  = (float*)   carve((size_t)T_TOK * HDIM * 4);
  float*    gbuf = (float*)   carve((size_t)T_TOK * ISHARED * 4);
  float*    ubuf = (float*)   carve((size_t)T_TOK * ISHARED * 4);
  _Float16* hbuf = (_Float16*)carve((size_t)T_TOK * ISHARED * 2);
  float*    wed  = (float*)   carve((size_t)T_TOK * NEXP * 4);

  const dim3 blk(256);

  // x = rmsnorm(hidden, w_in_ln)
  rmsnorm_kernel<<<T_TOK, blk, 0, stream>>>(hidden, w_in_ln, x1h, nullptr);

  // q/k/v projections
  launch_gemm(dim3((NHEAD * DHEAD) / BN, T_TOK / BM), stream,
              x1h, wq, qf32, T_TOK, NHEAD * DHEAD, HDIM, nullptr, nullptr, 0);
  launch_gemm(dim3((NKVH * DHEAD) / BN, T_TOK / BM), stream,
              x1h, wk, kf32, T_TOK, NKVH * DHEAD, HDIM, nullptr, nullptr, 0);
  launch_gemm(dim3((NKVH * DHEAD) / BN, T_TOK / BM), stream,
              x1h, wv, vf32, T_TOK, NKVH * DHEAD, HDIM, nullptr, nullptr, 0);

  // rope + f16 convert
  rope_cvt_kernel<<<T_TOK, blk, 0, stream>>>(positions, qf32, kf32, vf32,
                                             q16, k16, v16);

  // fused causal attention
  const size_t attn_smem = (size_t)16 * T_TOK * 4   // scores f32
                         + (size_t)16 * T_TOK * 2   // exp f16
                         + (256 + 16) * 4;          // reductions
  attn_kernel<<<dim3(T_TOK / 16, NHEAD), blk, attn_smem, stream>>>(q16, k16, v16, o16);

  // residual = hidden + o @ wo
  launch_gemm(dim3(HDIM / BN, T_TOK / BM), stream,
              o16, wo, resid, T_TOK, HDIM, NHEAD * DHEAD, hidden, nullptr, 0);

  // x2 = rmsnorm(residual, w_post_ln)
  rmsnorm_kernel<<<T_TOK, blk, 0, stream>>>(resid, w_post_ln, x2h, x2f);

  // shared expert: out = (silu(x2 @ ws_gate) * (x2 @ ws_up)) @ ws_down
  launch_gemm(dim3(ISHARED / BN, T_TOK / BM), stream,
              x2h, ws_gate, gbuf, T_TOK, ISHARED, HDIM, nullptr, nullptr, 0);
  launch_gemm(dim3(ISHARED / BN, T_TOK / BM), stream,
              x2h, ws_up, ubuf, T_TOK, ISHARED, HDIM, nullptr, nullptr, 0);
  silu_mul_kernel<<<(T_TOK * ISHARED + 255) / 256, blk, 0, stream>>>(
      gbuf, ubuf, hbuf, T_TOK * ISHARED);
  launch_gemm(dim3(HDIM / BN, T_TOK / BM), stream,
              hbuf, ws_down, out, T_TOK, HDIM, ISHARED, nullptr, nullptr, 0);

  // router weights
  router_kernel<<<T_TOK, blk, 0, stream>>>(x2f, gate_w, gate_b, wed);

  // MoE dense scan (matches reference): out += we[:,e] * expert_e(x2)
  for (int e = 0; e < NEXP; ++e) {
    const float* wg = we_gate + (size_t)e * HDIM * IEXP;
    const float* wu = we_up   + (size_t)e * HDIM * IEXP;
    const float* wd = we_down + (size_t)e * IEXP * HDIM;
    launch_gemm(dim3(IEXP / BN, T_TOK / BM), stream,
                x2h, wg, gbuf, T_TOK, IEXP, HDIM, nullptr, nullptr, 0);
    launch_gemm(dim3(IEXP / BN, T_TOK / BM), stream,
                x2h, wu, ubuf, T_TOK, IEXP, HDIM, nullptr, nullptr, 0);
    silu_mul_kernel<<<(T_TOK * IEXP + 255) / 256, blk, 0, stream>>>(
        gbuf, ubuf, hbuf, T_TOK * IEXP);
    launch_gemm(dim3(HDIM / BN, T_TOK / BM), stream,
                hbuf, wd, out, T_TOK, HDIM, IEXP, nullptr, wed + e, NEXP);
  }
}
